// LinearModel_57208964383148
// MI455X (gfx1250) — 8-bit matrix operand order repaired and checked by reading, compiled, not run on hardware
//
#include <hip/hip_runtime.h>
#include <cstdint>
#include <limits.h>

typedef __attribute__((ext_vector_type(8))) int v8i;

#define BATCH 65536
#define K1    784
#define K1P   832
#define HDIM  128
#define ODIM  10
#define ODIMP 16

#if __has_builtin(__builtin_amdgcn_global_load_async_to_lds_b128) && \
    __has_builtin(__builtin_amdgcn_s_wait_asynccnt)
#define USE_ASYNC_LDS 1
typedef int v4i_n __attribute__((vector_size(16)));
typedef v4i_n __attribute__((address_space(1))) as1_v4i;
typedef v4i_n __attribute__((address_space(3))) as3_v4i;
#else
#define USE_ASYNC_LDS 0
#endif

__device__ __forceinline__ int quant_clip(float r) {
    int v = (int)rintf(r);
    v = v < -128 ? -128 : v;
    v = v >  127 ?  127 : v;
    return v;
}

__global__ void k_reset(double* colSum, double* colSumSq, unsigned* amaxBits, int* qmax) {
    int t = threadIdx.x;
    if (t < 128)       { colSum[t] = 0.0; colSumSq[t] = 0.0; }
    else if (t == 128) { *amaxBits = 0u; }
    else if (t == 129) { *qmax = INT_MIN; }
}

__global__ void k_amax_f32(const float* __restrict__ x, size_t n,
                           unsigned* __restrict__ amaxBits) {
    __shared__ unsigned sm;
    if (threadIdx.x == 0) sm = 0u;
    __syncthreads();
    float m = 0.f;
    for (size_t i = (size_t)blockIdx.x * blockDim.x + threadIdx.x; i < n;
         i += (size_t)gridDim.x * blockDim.x)
        m = fmaxf(m, fabsf(x[i]));
    atomicMax(&sm, __float_as_uint(m));
    __syncthreads();
    if (threadIdx.x == 0) atomicMax(amaxBits, sm);
}

__global__ void k_quant_in(const float* __restrict__ x, int8_t* __restrict__ q,
                           const unsigned* __restrict__ amaxBits, float* __restrict__ sOut) {
    float s = fmaxf(__uint_as_float(*amaxBits), 1e-8f) * (1.f / 127.f);
    if (blockIdx.x == 0 && threadIdx.x == 0) *sOut = s;
    size_t n = (size_t)BATCH * K1P;
    for (size_t i = (size_t)blockIdx.x * blockDim.x + threadIdx.x; i < n;
         i += (size_t)gridDim.x * blockDim.x) {
        size_t row = i / K1P;
        int    col = (int)(i - row * (size_t)K1P);
        int qv = 0;
        if (col < K1) qv = quant_clip(x[row * (size_t)K1 + col] / s);
        q[i] = (int8_t)qv;
    }
}

__global__ void k_pack_w(const float* __restrict__ W, int rows, int K, int Kpad,
                         int8_t* __restrict__ out, size_t n) {
    for (size_t i = (size_t)blockIdx.x * blockDim.x + threadIdx.x; i < n;
         i += (size_t)gridDim.x * blockDim.x) {
        int r = (int)(i / Kpad);
        int c = (int)(i - (size_t)r * Kpad);
        int8_t v = 0;
        if (r < rows && c < K) v = (W[(size_t)r * K + c] >= 0.f) ? (int8_t)1 : (int8_t)-1;
        out[i] = v;
    }
}

template <int NT>
__global__ __launch_bounds__(256)
void k_gemm(const int8_t* __restrict__ qA, const int8_t* __restrict__ Wp, int Kpad,
            int* __restrict__ accOut, double* __restrict__ colSum,
            double* __restrict__ colSumSq) {
    extern __shared__ char smem[];
    int8_t* wl   = (int8_t*)smem;
    float* sstat = (float*)(smem + (size_t)NT * 16 * Kpad);
    const int tid    = threadIdx.x;
    const int wBytes = NT * 16 * Kpad;
    const int N      = NT * 16;

#if USE_ASYNC_LDS
    for (int off = tid * 16; off < wBytes; off += 256 * 16) {
        as1_v4i* gsrc = (as1_v4i*)(uintptr_t)(Wp + off);
        as3_v4i* ldst = (as3_v4i*)(uint32_t)(uintptr_t)(wl + off);
        __builtin_amdgcn_global_load_async_to_lds_b128(gsrc, ldst, 0, 0);
    }
    __builtin_amdgcn_s_wait_asynccnt(0);
#else
    for (int off = tid * 16; off < wBytes; off += 256 * 16)
        *(int4*)(wl + off) = *(const int4*)(Wp + off);
#endif
    for (int i = tid; i < 2 * N; i += 256) sstat[i] = 0.f;
    __syncthreads();

    const int lane  = tid & 31;
    const int wave  = tid >> 5;
    const int mBase = blockIdx.x * 128 + wave * 16;
    const int mrow = mBase + (lane & 15);
    const uint8_t* ap = (const uint8_t*)qA + (size_t)mrow * (size_t)Kpad + ((lane & 16) ? 16 : 0);
    const int colB  = lane & 15;
    const int koffB = (lane & 16) ? 16 : 0;

    v8i acc[NT];
#pragma unroll
    for (int t = 0; t < NT; ++t) acc[t] = (v8i){0, 0, 0, 0, 0, 0, 0, 0};

    for (int kc = 0; kc < Kpad; kc += 64) {
        int4 d0 = *(const int4*)(ap + kc);
        int4 d1 = *(const int4*)(ap + kc + 32);
        v8i a = {d0.x, d0.y, d0.z, d0.w, d1.x, d1.y, d1.z, d1.w};
#pragma unroll
        for (int t = 0; t < NT; ++t) {
            const int8_t* bp = wl + (size_t)(t * 16 + colB) * Kpad + kc + koffB;
            int4 b0 = *(const int4*)bp;
            int4 b1 = *(const int4*)(bp + 32);
            v8i b = {b0.x, b0.y, b0.z, b0.w, b1.x, b1.y, b1.z, b1.w};
            acc[t] = __builtin_amdgcn_wmma_i32_16x16x64_iu8(
                true, a, true, b, acc[t], false, false);
        }
    }

    const int mHi = (lane & 16) ? 8 : 0;
#pragma unroll
    for (int t = 0; t < NT; ++t) {
        float ls = 0.f, lss = 0.f;
#pragma unroll
        for (int r = 0; r < 8; ++r) {
            int v   = acc[t][r];
            int row = mBase + mHi + r;
            accOut[(size_t)row * N + t * 16 + colB] = v;
            float f = (float)v;
            ls += f;
            lss += f * f;
        }
        atomicAdd(&sstat[t * 16 + colB], ls);
        atomicAdd(&sstat[N + t * 16 + colB], lss);
    }
    __syncthreads();
    if (tid < N) {
        atomicAdd(&colSum[tid],   (double)sstat[tid]);
        atomicAdd(&colSumSq[tid], (double)sstat[N + tid]);
    }
}

__global__ void k_bn_prep(const double* __restrict__ colSum, const double* __restrict__ colSumSq,
                          const float* __restrict__ gamma, const float* __restrict__ beta,
                          const float* __restrict__ sIn, int ncols,
                          float* __restrict__ Acf, float* __restrict__ Ccf) {
    int c = threadIdx.x;
    if (c < ncols) {
        double s  = (double)(*sIn);
        double ma = colSum[c] * (1.0 / BATCH);
        double va = colSumSq[c] * (1.0 / BATCH) - ma * ma;
        float v    = (float)(s * s * va);
        float rinv = rsqrtf(v + 1e-5f);
        float A = gamma[c] * (float)s * rinv;
        Acf[c] = A;
        Ccf[c] = beta[c] - A * (float)ma;
    }
}

__global__ void k_amax_h(const int* __restrict__ acc, const float* __restrict__ Acf,
                         const float* __restrict__ Ccf, int ncols, int stride,
                         unsigned* __restrict__ amaxBits) {
    __shared__ unsigned sm;
    if (threadIdx.x == 0) sm = 0u;
    __syncthreads();
    float m = 0.f;
    size_t n = (size_t)BATCH * ncols;
    for (size_t i = (size_t)blockIdx.x * blockDim.x + threadIdx.x; i < n;
         i += (size_t)gridDim.x * blockDim.x) {
        size_t row = i / ncols;
        int    c   = (int)(i - row * (size_t)ncols);
        float  h   = Acf[c] * (float)acc[row * (size_t)stride + c] + Ccf[c];
        m = fmaxf(m, fabsf(h));
    }
    atomicMax(&sm, __float_as_uint(m));
    __syncthreads();
    if (threadIdx.x == 0) atomicMax(amaxBits, sm);
}

__global__ void k_qmax(const int* __restrict__ acc, const float* __restrict__ Acf,
                       const float* __restrict__ Ccf, int ncols, int stride,
                       const unsigned* __restrict__ amaxBits, float* __restrict__ sBnOut,
                       int* __restrict__ qmaxOut) {
    float s1 = fmaxf(__uint_as_float(*amaxBits), 1e-8f) * (1.f / 127.f);
    if (blockIdx.x == 0 && threadIdx.x == 0) *sBnOut = s1;
    __shared__ int sm;
    if (threadIdx.x == 0) sm = INT_MIN;
    __syncthreads();
    int mq = INT_MIN;
    size_t n = (size_t)BATCH * ncols;
    for (size_t i = (size_t)blockIdx.x * blockDim.x + threadIdx.x; i < n;
         i += (size_t)gridDim.x * blockDim.x) {
        size_t row = i / ncols;
        int    c   = (int)(i - row * (size_t)ncols);
        float  h   = Acf[c] * (float)acc[row * (size_t)stride + c] + Ccf[c];
        int q = quant_clip(h / s1);
        mq = q > mq ? q : mq;
    }
    atomicMax(&sm, mq);
    __syncthreads();
    if (threadIdx.x == 0) atomicMax(qmaxOut, sm);
}

__global__ void k_finalize(const int* __restrict__ acc, const float* __restrict__ Acf,
                           const float* __restrict__ Ccf, const float* __restrict__ sBnIn,
                           const int* __restrict__ qmaxIn, int actType  ,
                           int ncols, int stride, int8_t* __restrict__ qout,
                           float* __restrict__ out1, float* __restrict__ out2,
                           float* __restrict__ sActOut) {
    float s1 = *sBnIn;
    int   qm = *qmaxIn;
    float amax2 = (actType == 0) ? s1 * (float)(qm > 0 ? qm : 0)
                                 : 1.f / (1.f + expf(-s1 * (float)qm));
    float s2 = fmaxf(amax2, 1e-8f) * (1.f / 127.f);
    if (blockIdx.x == 0 && threadIdx.x == 0) *sActOut = s2;
    size_t n = (size_t)BATCH * ncols;
    for (size_t i = (size_t)blockIdx.x * blockDim.x + threadIdx.x; i < n;
         i += (size_t)gridDim.x * blockDim.x) {
        size_t row = i / ncols;
        int    c   = (int)(i - row * (size_t)ncols);
        float  h   = Acf[c] * (float)acc[row * (size_t)stride + c] + Ccf[c];
        int   q1 = quant_clip(h / s1);
        float hq = (float)q1 * s1;
        float a  = (actType == 0) ? fmaxf(hq, 0.f) : 1.f / (1.f + expf(-hq));
        int   q2 = quant_clip(a / s2);
        float o  = (float)q2 * s2;
        out1[i] = o;
        if (out2) out2[i] = o;
        if (qout) qout[i] = (int8_t)q2;
    }
}

extern "C" void kernel_launch(void* const* d_in, const int* in_sizes, int n_in,
                              void* d_out, int out_size, void* d_ws, size_t ws_size,
                              hipStream_t stream) {
    (void)in_sizes; (void)n_in; (void)out_size; (void)ws_size;

    const float* x   = (const float*)d_in[0];
    const float* W1  = (const float*)d_in[1];
    const float* g1  = (const float*)d_in[3];
    const float* be1 = (const float*)d_in[4];
    const float* W2  = (const float*)d_in[5];
    const float* g2  = (const float*)d_in[7];
    const float* be2 = (const float*)d_in[8];
    const float* W3  = (const float*)d_in[9];
    const float* g3  = (const float*)d_in[11];
    const float* be3 = (const float*)d_in[12];

    uint8_t* ws = (uint8_t*)d_ws;
    size_t off = 0;
    auto carve = [&](size_t bytes) { size_t o = off; off = (off + bytes + 255) & ~(size_t)255; return o; };
    int8_t*   qIn      = (int8_t*)(ws + carve((size_t)BATCH * K1P));
    int*      accBuf   = (int*)   (ws + carve((size_t)BATCH * HDIM * 4));
    int8_t*   qH1      = (int8_t*)(ws + carve((size_t)BATCH * HDIM));
    int8_t*   qH2      = (int8_t*)(ws + carve((size_t)BATCH * HDIM));
    int8_t*   Wp1      = (int8_t*)(ws + carve((size_t)HDIM * K1P));
    int8_t*   Wp2      = (int8_t*)(ws + carve((size_t)HDIM * HDIM));
    int8_t*   Wp3      = (int8_t*)(ws + carve((size_t)ODIMP * HDIM));
    double*   colSum   = (double*)(ws + carve(128 * sizeof(double)));
    double*   colSumSq = (double*)(ws + carve(128 * sizeof(double)));
    float*    Acf      = (float*) (ws + carve(128 * sizeof(float)));
    float*    Ccf      = (float*) (ws + carve(128 * sizeof(float)));
    unsigned* amaxBits = (unsigned*)(ws + carve(sizeof(unsigned)));
    float*    sBn      = (float*) (ws + carve(sizeof(float)));
    int*      qmax     = (int*)   (ws + carve(sizeof(int)));
    float*    sAct     = (float*) (ws + carve(4 * sizeof(float)));

    float* out    = (float*)d_out;
    float* out_h3a = out;
    float* out_h1  = out_h3a + (size_t)BATCH * ODIM;
    float* out_h2  = out_h1  + (size_t)BATCH * HDIM;
    float* out_h3b = out_h2  + (size_t)BATCH * HDIM;

    const int EB = 2048;
    dim3 blk(256);

    k_reset<<<1, 256, 0, stream>>>(colSum, colSumSq, amaxBits, qmax);
    k_amax_f32<<<EB, blk, 0, stream>>>(x, (size_t)BATCH * K1, amaxBits);
    k_quant_in<<<EB, blk, 0, stream>>>(x, qIn, amaxBits, &sAct[0]);
    k_pack_w<<<416, blk, 0, stream>>>(W1, HDIM, K1, K1P, Wp1, (size_t)HDIM * K1P);
    k_pack_w<<<64, blk, 0, stream>>>(W2, HDIM, HDIM, HDIM, Wp2, (size_t)HDIM * HDIM);
    k_pack_w<<<8, blk, 0, stream>>>(W3, ODIM, HDIM, HDIM, Wp3, (size_t)ODIMP * HDIM);

    k_reset<<<1, 256, 0, stream>>>(colSum, colSumSq, amaxBits, qmax);
    k_gemm<8><<<BATCH / 128, blk, (size_t)HDIM * K1P + 2 * 128 * 4, stream>>>(
        qIn, Wp1, K1P, accBuf, colSum, colSumSq);
    k_bn_prep<<<1, 128, 0, stream>>>(colSum, colSumSq, g1, be1, &sAct[0], HDIM, Acf, Ccf);
    k_amax_h<<<EB, blk, 0, stream>>>(accBuf, Acf, Ccf, HDIM, HDIM, amaxBits);
    k_qmax<<<EB, blk, 0, stream>>>(accBuf, Acf, Ccf, HDIM, HDIM, amaxBits, sBn, qmax);
    k_finalize<<<EB, blk, 0, stream>>>(accBuf, Acf, Ccf, sBn, qmax, 0, HDIM, HDIM,
                                       qH1, out_h1, nullptr, &sAct[1]);

    k_reset<<<1, 256, 0, stream>>>(colSum, colSumSq, amaxBits, qmax);
    k_gemm<8><<<BATCH / 128, blk, (size_t)HDIM * HDIM + 2 * 128 * 4, stream>>>(
        qH1, Wp2, HDIM, accBuf, colSum, colSumSq);
    k_bn_prep<<<1, 128, 0, stream>>>(colSum, colSumSq, g2, be2, &sAct[1], HDIM, Acf, Ccf);
    k_amax_h<<<EB, blk, 0, stream>>>(accBuf, Acf, Ccf, HDIM, HDIM, amaxBits);
    k_qmax<<<EB, blk, 0, stream>>>(accBuf, Acf, Ccf, HDIM, HDIM, amaxBits, sBn, qmax);
    k_finalize<<<EB, blk, 0, stream>>>(accBuf, Acf, Ccf, sBn, qmax, 0, HDIM, HDIM,
                                       qH2, out_h2, nullptr, &sAct[2]);

    k_reset<<<1, 256, 0, stream>>>(colSum, colSumSq, amaxBits, qmax);
    k_gemm<1><<<BATCH / 128, blk, (size_t)ODIMP * HDIM + 2 * 16 * 4, stream>>>(
        qH2, Wp3, HDIM, accBuf, colSum, colSumSq);
    k_bn_prep<<<1, 32, 0, stream>>>(colSum, colSumSq, g3, be3, &sAct[2], ODIM, Acf, Ccf);
    k_amax_h<<<EB, blk, 0, stream>>>(accBuf, Acf, Ccf, ODIM, ODIMP, amaxBits);
    k_qmax<<<EB, blk, 0, stream>>>(accBuf, Acf, Ccf, ODIM, ODIMP, amaxBits, sBn, qmax);
    k_finalize<<<EB, blk, 0, stream>>>(accBuf, Acf, Ccf, sBn, qmax, 1, ODIM, ODIMP,
                                       nullptr, out_h3a, out_h3b, &sAct[3]);
}
